// PSA_76673756168617
// MI455X (gfx1250) — matrix operand order repaired and checked by reading, not compiled, not run on hardware
//
#include <hip/hip_runtime.h>

typedef __attribute__((ext_vector_type(16))) _Float16 v16h;
typedef __attribute__((ext_vector_type(8)))  float    v8f;
typedef __attribute__((ext_vector_type(4)))  int      v4i;

// ---------------------------------------------------------------------------
// Optional CDNA5 async global->LDS copies (ASYNCcnt), compile-guarded.
// Param types per hipcc diagnostic: (v4i AS1*, v4i AS3*, imm, imm)
// ---------------------------------------------------------------------------
#if defined(__has_builtin)
#if __has_builtin(__builtin_amdgcn_global_load_async_to_lds_b128) && \
    __has_builtin(__builtin_amdgcn_s_wait_asynccnt)
#define USE_ASYNC 1
#endif
#endif
#ifndef USE_ASYNC
#define USE_ASYNC 0
#endif

#if USE_ASYNC
__device__ __forceinline__ void async_copy16(const void* g, void* l) {
    __builtin_amdgcn_global_load_async_to_lds_b128(
        (__attribute__((address_space(1))) v4i*)(g),
        (__attribute__((address_space(3))) v4i*)(l), 0, 0);
}
__device__ __forceinline__ void async_wait() { __builtin_amdgcn_s_wait_asynccnt(0); }
#else
__device__ __forceinline__ void async_copy16(const void* g, void* l) {
    *(uint4*)l = *(const uint4*)g;
}
__device__ __forceinline__ void async_wait() {}
#endif

// ---------------------------------------------------------------------------
// WMMA fragment loaders from LDS tiles stored K-contiguous per row/column.
// A operand: VGPR r holds K pair k = (r<4?0:16) + khalf*8 + (r&3)*2
// B operand: VGPR r holds K pair k = (r<4?0:16) + khalf*8 + (r&3)*2 (same as A)
// ---------------------------------------------------------------------------
__device__ __forceinline__ v16h frag_a(const _Float16* rowbase, int khalf, int koff) {
    union { v16h v; uint32_t u[8]; } f;
    const uint32_t* p = (const uint32_t*)rowbase;
#pragma unroll
    for (int r = 0; r < 8; ++r) {
        int k = ((r < 4) ? 0 : 16) + khalf * 8 + (r & 3) * 2 + koff;
        f.u[r] = p[k >> 1];
    }
    return f.v;
}

__device__ __forceinline__ v16h frag_b(const _Float16* colbase, int khalf, int koff) {
    union { v16h v; uint32_t u[8]; } f;
    const uint32_t* p = (const uint32_t*)colbase + ((khalf * 8 + koff) >> 1);
#pragma unroll
    for (int r = 0; r < 8; ++r) f.u[r] = p[(r < 4) ? r : r + 4];
    return f.v;
}

#define WMMA_F16F32(a, b, c) \
    __builtin_amdgcn_wmma_f32_16x16x32_f16(false, (a), false, (b), (short)0, (c), false, false)

// ---------------------------------------------------------------------------
// Fused 1x1-conv GEMM: Out_b (Cout x 1024) = W (Cout x Cin) * X_b, + BN/ReLU.
// Weights pre-converted to f16 (async-staged); activations InT = f32 or f16.
// Block = 128 threads (4 waves): 64(M) x 64(N) tile, wave owns 16 M-rows.
// ---------------------------------------------------------------------------
template <typename InT, typename OutT>
__global__ __launch_bounds__(128) void conv1x1_wmma(
    const InT* __restrict__ in, int b_stride_in, int Cin,
    const _Float16* __restrict__ W16,
    OutT* __restrict__ out, int b_stride_out,
    const float* __restrict__ gamma, const float* __restrict__ beta, int relu)
{
    __shared__ alignas(16) _Float16 lsA[64 * 32];   // [m][k]
    __shared__ alignas(16) _Float16 lsB[64 * 32];   // [n][k]

    const int b   = blockIdx.z;
    const int co0 = blockIdx.x * 64;
    const int p0  = blockIdx.y * 64;
    const int tid = threadIdx.x;
    const int lane = tid & 31, wave = tid >> 5;
    const int khalf = lane >> 4, lidx = lane & 15;

    const InT* inb = in + (size_t)b * b_stride_in;

    v8f acc[4] = {};

    for (int k0 = 0; k0 < Cin; k0 += 32) {
        // A tile: f16 weights, rows contiguous -> async 16B copies (2/thread)
        for (int t = tid; t < 256; t += 128) {
            int m = t >> 2, c = t & 3;
            async_copy16(W16 + (size_t)(co0 + m) * Cin + k0 + c * 8,
                         &lsA[m * 32 + c * 8]);
        }
        // B tile: transpose staging, 16B vector global loads along p
        constexpr int VEC = 16 / (int)sizeof(InT);   // 4 (f32) or 8 (f16)
        constexpr int PER_ROW = 64 / VEC;
        for (int t = tid; t < 32 * PER_ROW; t += 128) {
            int k = t / PER_ROW, nv = t % PER_ROW;
            union { uint4 u; InT e[VEC]; } v;
            v.u = *(const uint4*)(inb + (size_t)(k0 + k) * 1024 + p0 + nv * VEC);
#pragma unroll
            for (int i = 0; i < VEC; ++i)
                lsB[(nv * VEC + i) * 32 + k] = (_Float16)v.e[i];
        }
        async_wait();
        __syncthreads();

        v16h a = frag_a(&lsA[(wave * 16 + lidx) * 32], khalf, 0);
#pragma unroll
        for (int nf = 0; nf < 4; ++nf) {
            v16h bf = frag_b(&lsB[(nf * 16 + lidx) * 32], khalf, 0);
            acc[nf] = WMMA_F16F32(a, bf, acc[nf]);
        }
        __syncthreads();
    }

    // Epilogue: lane holds col N=lidx, rows M = khalf*8 + r
    OutT* outb = out + (size_t)b * b_stride_out;
#pragma unroll
    for (int nf = 0; nf < 4; ++nf) {
        int p = p0 + nf * 16 + lidx;
#pragma unroll
        for (int r = 0; r < 8; ++r) {
            int co = co0 + wave * 16 + khalf * 8 + r;
            float v = acc[nf][r];
            if (gamma) v = v * gamma[co] + beta[co];
            if (relu)  v = fmaxf(v, 0.0f);
            outb[(size_t)co * 1024 + p] = (OutT)v;
        }
    }
}

// ---------------------------------------------------------------------------
// Flash attention per (batch, head, 64-query tile).  d=16 (padded K=32),
// seq=1024, all operands f16 in global.  4 waves; wave owns 16 query rows.
// qkv channels per head: [q:0-15 | k:16-31 | v:32-47] at base h*48.
// ---------------------------------------------------------------------------
__global__ __launch_bounds__(128) void qattn_wmma(
    const _Float16* __restrict__ qkv, _Float16* __restrict__ attn_out)
{
    __shared__ alignas(16) _Float16 lsQ[64 * 32];      // [i][d(pad32)]
    __shared__ alignas(16) _Float16 lsK[64 * 32];      // [j][d(pad32)]
    __shared__ alignas(16) _Float16 lsV[16 * 64];      // [d][j]
    __shared__ alignas(16) _Float16 lsP[4][16 * 64];   // per-wave [i_local][j]
    __shared__ float lsRow[4][16];

    const int b = blockIdx.z, h = blockIdx.y;
    const int i0 = blockIdx.x * 64;
    const int tid = threadIdx.x;
    const int lane = tid & 31, wave = tid >> 5;
    const int khalf = lane >> 4, lidx = lane & 15;

    const _Float16* qg = qkv + ((size_t)b * 384 + h * 48) * 1024;
    const _Float16* kg = qg + 16 * 1024;
    const _Float16* vg = qg + 32 * 1024;

    // zero-pad d=16..31 for Q and K tiles (disjoint from data region)
    for (int t = tid; t < 64 * 16; t += 128) {
        int i = t >> 4, d = 16 + (t & 15);
        lsQ[i * 32 + d] = (_Float16)0.0f;
        lsK[i * 32 + d] = (_Float16)0.0f;
    }
    // Q tile: transpose staging, 8-half vector reads along i (1 op/thread)
    for (int t = tid; t < 16 * 8; t += 128) {
        int d = t >> 3, iv = t & 7;
        union { uint4 u; _Float16 e[8]; } v;
        v.u = *(const uint4*)(qg + (size_t)d * 1024 + i0 + iv * 8);
#pragma unroll
        for (int i = 0; i < 8; ++i) lsQ[(iv * 8 + i) * 32 + d] = v.e[i];
    }

    float mrow[8], lrow[8];
#pragma unroll
    for (int r = 0; r < 8; ++r) { mrow[r] = -1e30f; lrow[r] = 0.0f; }
    v8f o = {};

    for (int j0 = 0; j0 < 1024; j0 += 64) {
        // V tile: contiguous -> async 16B copies (1 op/thread)
        for (int t = tid; t < 128; t += 128) {
            int d = t >> 3, c = t & 7;
            async_copy16(vg + (size_t)d * 1024 + j0 + c * 8, &lsV[d * 64 + c * 8]);
        }
        // K tile: transpose staging
        for (int t = tid; t < 16 * 8; t += 128) {
            int d = t >> 3, jv = t & 7;
            union { uint4 u; _Float16 e[8]; } v;
            v.u = *(const uint4*)(kg + (size_t)d * 1024 + j0 + jv * 8);
#pragma unroll
            for (int j = 0; j < 8; ++j) lsK[(jv * 8 + j) * 32 + d] = v.e[j];
        }
        async_wait();
        __syncthreads();

        // S = (Q^T K) * scale
        v16h aq = frag_a(&lsQ[(wave * 16 + lidx) * 32], khalf, 0);
        v8f s[4];
#pragma unroll
        for (int nf = 0; nf < 4; ++nf) {
            v16h bk = frag_b(&lsK[(nf * 16 + lidx) * 32], khalf, 0);
            v8f z = {};
            s[nf] = WMMA_F16F32(aq, bk, z);
        }

        // online softmax (rows khalf*8+r, redundant over 16 lanes)
        float corr[8];
#pragma unroll
        for (int r = 0; r < 8; ++r) {
            float mx = -1e30f;
#pragma unroll
            for (int nf = 0; nf < 4; ++nf) { s[nf][r] *= 0.25f; mx = fmaxf(mx, s[nf][r]); }
            for (int off = 1; off < 16; off <<= 1) mx = fmaxf(mx, __shfl_xor(mx, off, 32));
            float mnew = fmaxf(mrow[r], mx);
            corr[r] = __expf(mrow[r] - mnew);
            mrow[r] = mnew;
            float sum = 0.0f;
#pragma unroll
            for (int nf = 0; nf < 4; ++nf) {
                float pe = __expf(s[nf][r] - mnew);
                s[nf][r] = pe;
                sum += pe;
            }
            for (int off = 1; off < 16; off <<= 1) sum += __shfl_xor(sum, off, 32);
            lrow[r] = lrow[r] * corr[r] + sum;
        }

        // P -> LDS [i_local][j]
#pragma unroll
        for (int nf = 0; nf < 4; ++nf)
#pragma unroll
            for (int r = 0; r < 8; ++r)
                lsP[wave][(khalf * 8 + r) * 64 + nf * 16 + lidx] = (_Float16)s[nf][r];
        if (lidx == 0) {
#pragma unroll
            for (int r = 0; r < 8; ++r) lsRow[wave][khalf * 8 + r] = corr[r];
        }
        __syncthreads();

        float ci = lsRow[wave][lidx];
#pragma unroll
        for (int r = 0; r < 8; ++r) o[r] *= ci;

        // O += V * P^T  (K=j in two 32-chunks)
        v16h av0 = frag_a(&lsV[lidx * 64], khalf, 0);
        v16h bp0 = frag_b(&lsP[wave][lidx * 64], khalf, 0);
        o = WMMA_F16F32(av0, bp0, o);
        v16h av1 = frag_a(&lsV[lidx * 64], khalf, 32);
        v16h bp1 = frag_b(&lsP[wave][lidx * 64], khalf, 32);
        o = WMMA_F16F32(av1, bp1, o);
        __syncthreads();
    }

    if (lidx == 0) {
#pragma unroll
        for (int r = 0; r < 8; ++r) lsRow[wave][khalf * 8 + r] = lrow[r];
    }
    __syncthreads();
    float inv = 1.0f / lsRow[wave][lidx];
    _Float16* og = attn_out + ((size_t)b * 128 + h * 16) * 1024;
    int i = i0 + wave * 16 + lidx;
#pragma unroll
    for (int r = 0; r < 8; ++r) {
        int d = khalf * 8 + r;
        og[(size_t)d * 1024 + i] = (_Float16)(o[r] * inv);
    }
}

// ---------------------------------------------------------------------------
// Depthwise 3x3 (pad 1) on v (f16, from qkv buffer), added into attn_out.
// ---------------------------------------------------------------------------
__global__ __launch_bounds__(256) void dwconv3x3_add(
    const _Float16* __restrict__ qkv, const float* __restrict__ pw,
    _Float16* __restrict__ outb)
{
    int bc = blockIdx.x;
    int b = bc >> 7, c = bc & 127;
    int h = c >> 4, d = c & 15;
    const _Float16* v = qkv + ((size_t)b * 384 + h * 48 + 32 + d) * 1024;
    const float* w = pw + c * 9;
    _Float16* o = outb + ((size_t)b * 128 + c) * 1024;
    for (int p = threadIdx.x; p < 1024; p += blockDim.x) {
        int y = p >> 5, x = p & 31;
        float acc = 0.0f;
#pragma unroll
        for (int ky = 0; ky < 3; ++ky) {
            int iy = y + ky - 1;
            if (iy < 0 || iy > 31) continue;
#pragma unroll
            for (int kx = 0; kx < 3; ++kx) {
                int ix = x + kx - 1;
                if (ix < 0 || ix > 31) continue;
                acc += w[ky * 3 + kx] * (float)v[iy * 32 + ix];
            }
        }
        o[p] = (_Float16)((float)o[p] + acc);
    }
}

// concat: copy b_branch (channels 128..255 of cv1 out) into cat buffer (f16)
__global__ __launch_bounds__(256) void copy_branch(
    const _Float16* __restrict__ y, _Float16* __restrict__ cat)
{
    size_t idx = ((size_t)blockIdx.x * 256 + threadIdx.x) * 8;   // halves
    if (idx < (size_t)8 * 128 * 1024) {
        size_t b = idx >> 17, rem = idx & 0x1FFFF;
        *(uint4*)&cat[b * 256 * 1024 + 128 * 1024 + rem] =
            *(const uint4*)&y[b * 256 * 1024 + 128 * 1024 + rem];
    }
}

// convert f32 weights -> f16 (n multiple of 4)
__global__ __launch_bounds__(256) void cvt_w_f16(
    const float* __restrict__ s, _Float16* __restrict__ d, int n)
{
    int i = ((int)blockIdx.x * 256 + (int)threadIdx.x) * 4;
    if (i < n) {
        float4 v = *(const float4*)(s + i);
        d[i + 0] = (_Float16)v.x; d[i + 1] = (_Float16)v.y;
        d[i + 2] = (_Float16)v.z; d[i + 3] = (_Float16)v.w;
    }
}

// ---------------------------------------------------------------------------
extern "C" void kernel_launch(void* const* d_in, const int* in_sizes, int n_in,
                              void* d_out, int out_size, void* d_ws, size_t ws_size,
                              hipStream_t stream) {
    const float* x      = (const float*)d_in[0];
    const float* cv1_w  = (const float*)d_in[1];
    const float* bn1_g  = (const float*)d_in[2];
    const float* bn1_b  = (const float*)d_in[3];
    const float* qkv_w  = (const float*)d_in[4];
    const float* pe_w   = (const float*)d_in[5];
    const float* proj_w = (const float*)d_in[6];
    const float* ffn1_w = (const float*)d_in[7];
    const float* fbn1_g = (const float*)d_in[8];
    const float* fbn1_b = (const float*)d_in[9];
    const float* ffn2_w = (const float*)d_in[10];
    const float* fbn2_g = (const float*)d_in[11];
    const float* fbn2_b = (const float*)d_in[12];
    const float* cv2_w  = (const float*)d_in[13];
    const float* bn2_g  = (const float*)d_in[14];
    const float* bn2_b  = (const float*)d_in[15];
    float* out = (float*)d_out;

    // f16 workspace layout (offsets in halves, all 16B aligned)
    _Float16* hws = (_Float16*)d_ws;
    _Float16* w_cv1  = hws + 0;          // 65536
    _Float16* w_qkv  = hws + 65536;      // 49152
    _Float16* w_proj = hws + 114688;     // 16384
    _Float16* w_ffn1 = hws + 131072;     // 32768
    _Float16* w_ffn2 = hws + 163840;     // 32768
    _Float16* w_cv2  = hws + 196608;     // 65536
    _Float16* ybuf   = hws + 262144;     // 2097152  cv1 out (a | b_branch)
    _Float16* qkvb   = hws + 2359296;    // 3145728
    _Float16* obuf   = hws + 5505024;    // 1048576  attn out (+pe)
    _Float16* pbuf   = hws + 6553600;    // 1048576  proj out
    _Float16* fbuf   = hws + 7602176;    // 2097152  ffn1 out
    _Float16* cbuf   = hws + 9699328;    // 2097152  concat

    dim3 blk(128);
    // weight conversion
    cvt_w_f16<<<64, 256, 0, stream>>>(cv1_w,  w_cv1,  65536);
    cvt_w_f16<<<48, 256, 0, stream>>>(qkv_w,  w_qkv,  49152);
    cvt_w_f16<<<16, 256, 0, stream>>>(proj_w, w_proj, 16384);
    cvt_w_f16<<<32, 256, 0, stream>>>(ffn1_w, w_ffn1, 32768);
    cvt_w_f16<<<32, 256, 0, stream>>>(ffn2_w, w_ffn2, 32768);
    cvt_w_f16<<<64, 256, 0, stream>>>(cv2_w,  w_cv2,  65536);

    // cv1 -> BN -> ReLU  (256 -> 256), f32 in, f16 out
    conv1x1_wmma<float, _Float16><<<dim3(4, 16, 8), blk, 0, stream>>>(
        x, 256 * 1024, 256, w_cv1, ybuf, 256 * 1024, bn1_g, bn1_b, 1);
    // qkv (128 -> 384)
    conv1x1_wmma<_Float16, _Float16><<<dim3(6, 16, 8), blk, 0, stream>>>(
        ybuf, 256 * 1024, 128, w_qkv, qkvb, 384 * 1024, nullptr, nullptr, 0);
    // attention
    qattn_wmma<<<dim3(16, 8, 8), blk, 0, stream>>>(qkvb, obuf);
    // positional encoding: += dwconv3x3(v)
    dwconv3x3_add<<<dim3(8 * 128), dim3(256), 0, stream>>>(qkvb, pe_w, obuf);
    // proj (128 -> 128)
    conv1x1_wmma<_Float16, _Float16><<<dim3(2, 16, 8), blk, 0, stream>>>(
        obuf, 128 * 1024, 128, w_proj, pbuf, 128 * 1024, nullptr, nullptr, 0);
    // ffn1 -> BN -> ReLU (128 -> 256)
    conv1x1_wmma<_Float16, _Float16><<<dim3(4, 16, 8), blk, 0, stream>>>(
        pbuf, 128 * 1024, 128, w_ffn1, fbuf, 256 * 1024, fbn1_g, fbn1_b, 1);
    // ffn2 -> BN (256 -> 128) into cat channels 0..127
    conv1x1_wmma<_Float16, _Float16><<<dim3(2, 16, 8), blk, 0, stream>>>(
        fbuf, 256 * 1024, 256, w_ffn2, cbuf, 256 * 1024, fbn2_g, fbn2_b, 0);
    // concat: b_branch into cat channels 128..255
    copy_branch<<<512, 256, 0, stream>>>(ybuf, cbuf);
    // cv2 -> BN -> ReLU (256 -> 256), f16 in, f32 out
    conv1x1_wmma<_Float16, float><<<dim3(4, 16, 8), blk, 0, stream>>>(
        cbuf, 256 * 1024, 256, w_cv2, out, 256 * 1024, bn2_g, bn2_b, 1);
}
